// SANet_L3S2_57844619543036
// MI455X (gfx1250) — matrix operand order repaired and checked by reading, not compiled, not run on hardware
//
#include <hip/hip_runtime.h>

typedef __attribute__((ext_vector_type(16))) __bf16 v16bf;
typedef __attribute__((ext_vector_type(8)))  float  v8f;

union BFrag { unsigned u[8]; v16bf v; };

#define RPN_VOX 131072   // 32*64*64 voxels per channel of comb2
#define NVOX    13824    // 24^3 voxels per proposal
#define NCHUNK  108      // 13824 / 128
#define OFF_RD  393216   // rpn_deltas offset (floats)
#define OFF_CL  2752512  // rcnn_logits offset
#define OFF_CD  2752640  // rcnn_deltas offset

// ---- CDNA5 async LDS copy path (guarded so compile never breaks) ----------
#ifndef HAVE_ASYNC
#if defined(__has_builtin)
#if __has_builtin(__builtin_amdgcn_global_load_async_to_lds_b128)
#define HAVE_ASYNC 1
#endif
#endif
#endif
#ifndef HAVE_ASYNC
#define HAVE_ASYNC 0
#endif

#if HAVE_ASYNC
typedef int v4i __attribute__((vector_size(16)));
typedef __attribute__((address_space(1))) v4i gv4i;   // global
typedef __attribute__((address_space(3))) v4i lv4i;   // LDS

__device__ __forceinline__ void async_copy_b128(const float* gsrc, float* ldst) {
  __builtin_amdgcn_global_load_async_to_lds_b128((gv4i*)gsrc, (lv4i*)ldst, 0, 0);
}
__device__ __forceinline__ void wait_async_le2() {
#if defined(__has_builtin) && __has_builtin(__builtin_amdgcn_s_wait_asynccnt)
  __builtin_amdgcn_s_wait_asynccnt(2);
#else
  asm volatile("s_wait_asynccnt 0x2" ::: "memory");
#endif
}
__device__ __forceinline__ void wait_async_le0() {
#if defined(__has_builtin) && __has_builtin(__builtin_amdgcn_s_wait_asynccnt)
  __builtin_amdgcn_s_wait_asynccnt(0);
#else
  asm volatile("s_wait_asynccnt 0x0" ::: "memory");
#endif
}
__device__ __forceinline__ void fence_lds_reads() {
  asm volatile("s_wait_dscnt 0x0" ::: "memory");
}
#endif

__device__ __forceinline__ unsigned short f2bf(float f) {
  unsigned u = __builtin_bit_cast(unsigned, f);
  u += 0x7FFFu + ((u >> 16) & 1u);           // round-to-nearest-even
  return (unsigned short)(u >> 16);
}
__device__ __forceinline__ float bf2f(unsigned short s) {
  return __builtin_bit_cast(float, ((unsigned)s) << 16);
}
// base K of the bf16 pair held in A-fragment VGPR v for lane-half h (16-bit A 16x32 layout)
__device__ __forceinline__ int abk(int v, int h) {
  return (v < 4 ? 2 * v : 16 + 2 * (v - 4)) + 8 * h;
}
__device__ __forceinline__ v8f wmma_bf16(const BFrag& a, const BFrag& b, v8f c) {
  return __builtin_amdgcn_wmma_f32_16x16x32_bf16(false, a.v, false, b.v, (short)0, c, false, false);
}

// ---------------------------------------------------------------- zero init
__global__ void k_zero(float* __restrict__ p, int n) {
  int i = blockIdx.x * 256 + threadIdx.x;
  if (i < n) p[i] = 0.f;
}

// ---------------------------------------------------------------- RPN head
__global__ __launch_bounds__(256) void k_rpn(
    const float* __restrict__ comb2,
    const float* __restrict__ w_rpn,  const float* __restrict__ b_rpn,
    const float* __restrict__ w_rlog, const float* __restrict__ b_rlog,
    const float* __restrict__ w_rdel, const float* __restrict__ b_rdel,
    float* __restrict__ out) {
  __shared__ unsigned short xbf[128][64];   // comb2 tile [ch][vox]
  __shared__ unsigned short wbf[64][128];   // w_rpn [o][c]
  __shared__ unsigned short hbf[64][64];    // h [vox][och]
  __shared__ unsigned short wc[64][32];     // combined rlog|rdel [k][n]
  const int t = threadIdx.x;
  const int vox0 = blockIdx.x * 64;

  for (int i = t; i < 128 * 64; i += 256) {
    int c = i >> 6, vx = i & 63;
    xbf[c][vx] = f2bf(comb2[c * RPN_VOX + vox0 + vx]);
  }
  for (int i = t; i < 64 * 128; i += 256) {
    int o = i >> 7, c = i & 127;
    wbf[o][c] = f2bf(w_rpn[o * 128 + c]);
  }
  for (int i = t; i < 64 * 32; i += 256) {
    int k = i >> 5, n = i & 31;
    float v = 0.f;
    if (n < 3)       v = w_rlog[n * 64 + k];
    else if (n < 21) v = w_rdel[(n - 3) * 64 + k];
    wc[k][n] = f2bf(v);
  }
  __syncthreads();

  const int wave = t >> 5, lane = t & 31;
  const int m = lane & 15, hh = lane >> 4;

  // stage 1: h[o][vox] = relu(W x + b), 4 o-tiles x 4 vox-groups, 2 jobs/wave
  for (int j = 0; j < 2; ++j) {
    int job = wave + j * 8;
    int mt = job >> 2, ng = job & 3;
    v8f acc = {};
#pragma unroll
    for (int ks = 0; ks < 4; ++ks) {
      int k0 = ks * 32;
      BFrag a, b;
#pragma unroll
      for (int v = 0; v < 8; ++v) a.u[v] = *(const unsigned*)&wbf[mt * 16 + m][k0 + abk(v, hh)];
#pragma unroll
      for (int v = 0; v < 8; ++v) b.u[v] = (unsigned)xbf[k0 + abk(v, hh)][ng * 16 + m] | ((unsigned)xbf[k0 + abk(v, hh) + 1][ng * 16 + m] << 16);
      acc = wmma_bf16(a, b, acc);
    }
    int vx = ng * 16 + m;
#pragma unroll
    for (int r = 0; r < 8; ++r) {
      int o = mt * 16 + r + 8 * hh;
      float hv = acc[r] + b_rpn[o];
      hbf[vx][o] = f2bf(hv > 0.f ? hv : 0.f);
    }
  }
  __syncthreads();

  // stage 2: [64 vox] x [64 k] x [32 n (3 logit + 18 delta + pad)], 1 job/wave
  {
    int mtv = wave & 3, nt = wave >> 2;
    v8f acc = {};
#pragma unroll
    for (int ks = 0; ks < 2; ++ks) {
      int k0 = ks * 32;
      BFrag a, b;
#pragma unroll
      for (int v = 0; v < 8; ++v) a.u[v] = *(const unsigned*)&hbf[mtv * 16 + m][k0 + abk(v, hh)];
#pragma unroll
      for (int v = 0; v < 8; ++v) b.u[v] = (unsigned)wc[k0 + abk(v, hh)][nt * 16 + m] | ((unsigned)wc[k0 + abk(v, hh) + 1][nt * 16 + m] << 16);
      acc = wmma_bf16(a, b, acc);
    }
    int n = nt * 16 + m;
#pragma unroll
    for (int r = 0; r < 8; ++r) {
      int vg = vox0 + mtv * 16 + r + 8 * hh;
      if (n < 3)       out[vg * 3 + n] = acc[r] + b_rlog[n];
      else if (n < 21) out[OFF_RD + vg * 18 + (n - 3)] = acc[r] + b_rdel[n - 3];
    }
  }
}

// ------------------------------------------------- up2: upsample + conv1x1
__global__ __launch_bounds__(256) void k_up2(
    const float* __restrict__ comb2, const int* __restrict__ props,
    const float* __restrict__ w_up2, const float* __restrict__ b_up2,
    unsigned short* __restrict__ vbuf, float* __restrict__ stats1) {
  __shared__ unsigned short ut[128][128];   // upsampled input [ch][vox]
  __shared__ unsigned short wbf[64][128];
  __shared__ float sstat[64][2];
  const int t = threadIdx.x;
  const int p = blockIdx.x / NCHUNK, chunk = blockIdx.x % NCHUNK;
  const int vbase = chunk * 128;
  const int z0 = props[p * 7 + 1] >> 2, y0 = props[p * 7 + 2] >> 2, x0 = props[p * 7 + 3] >> 2;

  if (t < 64) { sstat[t][0] = 0.f; sstat[t][1] = 0.f; }
  for (int i = t; i < 64 * 128; i += 256) {
    int o = i >> 7, c = i & 127;
    wbf[o][c] = f2bf(w_up2[o * 128 + c]);
  }
  // trilinear (align_corners) 12 -> 24, direct 8-corner gather (L2-resident)
  {
    int vx = t & 127;
    int cbase = (t >> 7) * 64;
    int vg = vbase + vx;
    int Z = vg / 576, rem = vg % 576, Y = rem / 24, X = rem % 24;
    float sz = Z * (11.f / 23.f), sy = Y * (11.f / 23.f), sx = X * (11.f / 23.f);
    int zl = (int)sz, yl = (int)sy, xl = (int)sx;
    float wz = sz - zl, wy = sy - yl, wx = sx - xl;
    int oz = (zl < 11) ? 4096 : 0, oy = (yl < 11) ? 64 : 0, ox = (xl < 11) ? 1 : 0;
    int a000 = (z0 + zl) * 4096 + (y0 + yl) * 64 + (x0 + xl);
    float w00 = (1.f - wy) * (1.f - wx), w01 = (1.f - wy) * wx;
    float w10 = wy * (1.f - wx),         w11 = wy * wx;
    for (int c = 0; c < 64; ++c) {
      const float* base = comb2 + (size_t)(cbase + c) * RPN_VOX + a000;
      float lo = base[0] * w00 + base[ox] * w01 + base[oy] * w10 + base[oy + ox] * w11;
      float hi = base[oz] * w00 + base[oz + ox] * w01 + base[oz + oy] * w10 + base[oz + oy + ox] * w11;
      ut[cbase + c][vx] = f2bf(lo + wz * (hi - lo));
    }
  }
  __syncthreads();

  const int wave = t >> 5, lane = t & 31;
  const int m = lane & 15, hh = lane >> 4;
  const int mt = wave & 3, ngb = (wave >> 2) * 4;
  float lsum[8] = {}, lsq[8] = {};
  for (int jg = 0; jg < 4; ++jg) {
    int ng = ngb + jg;
    v8f acc = {};
#pragma unroll
    for (int ks = 0; ks < 4; ++ks) {
      int k0 = ks * 32;
      BFrag a, b;
#pragma unroll
      for (int v = 0; v < 8; ++v) a.u[v] = *(const unsigned*)&wbf[mt * 16 + m][k0 + abk(v, hh)];
#pragma unroll
      for (int v = 0; v < 8; ++v) b.u[v] = (unsigned)ut[k0 + abk(v, hh)][ng * 16 + m] | ((unsigned)ut[k0 + abk(v, hh) + 1][ng * 16 + m] << 16);
      acc = wmma_bf16(a, b, acc);
    }
    int vg = vbase + ng * 16 + m;
#pragma unroll
    for (int r = 0; r < 8; ++r) {
      int o = mt * 16 + r + 8 * hh;
      float val = acc[r] + b_up2[o];
      vbuf[((size_t)p * 64 + o) * NVOX + vg] = f2bf(val);
      lsum[r] += val; lsq[r] += val * val;
    }
  }
#pragma unroll
  for (int r = 0; r < 8; ++r) {
    int o = mt * 16 + r + 8 * hh;
    atomicAdd(&sstat[o][0], lsum[r]);
    atomicAdd(&sstat[o][1], lsq[r]);
  }
  __syncthreads();
  if (t < 64) {
    atomicAdd(stats1 + (p * 64 + t) * 2 + 0, sstat[t][0]);
    atomicAdd(stats1 + (p * 64 + t) * 2 + 1, sstat[t][1]);
  }
}

// -------------------------------------- back2: norm(v)|o1 concat + conv1x1
__global__ __launch_bounds__(256) void k_back2(
    const float* __restrict__ out1, const int* __restrict__ props,
    const float* __restrict__ w_back2, const float* __restrict__ b_back2,
    const float* __restrict__ g_up2, const float* __restrict__ be_up2,
    const float* __restrict__ stats1,
    unsigned short* __restrict__ vbuf, float* __restrict__ stats2) {
  __shared__ unsigned short xin[128][128];
  __shared__ unsigned short wbf[64][128];
  __shared__ float sc[64], shf[64];
  __shared__ float sstat[64][2];
  const int t = threadIdx.x;
  const int p = blockIdx.x / NCHUNK, chunk = blockIdx.x % NCHUNK;
  const int vbase = chunk * 128;
  const int z0 = props[p * 7 + 1] >> 1, y0 = props[p * 7 + 2] >> 1, x0 = props[p * 7 + 3] >> 1;

  if (t < 64) {
    float s = stats1[(p * 64 + t) * 2], sq = stats1[(p * 64 + t) * 2 + 1];
    float mu = s * (1.f / NVOX);
    float var = sq * (1.f / NVOX) - mu * mu;
    float rs = rsqrtf(var + 1e-5f);
    float scale = g_up2[t] * rs;
    sc[t] = scale; shf[t] = be_up2[t] - scale * mu;
    sstat[t][0] = 0.f; sstat[t][1] = 0.f;
  }
  for (int i = t; i < 64 * 128; i += 256) {
    int o = i >> 7, c = i & 127;
    wbf[o][c] = f2bf(w_back2[o * 128 + c]);
  }
  __syncthreads();
  {
    int vx = t & 127;
    int vg = vbase + vx;
    int Z = vg / 576, rem = vg % 576, Y = rem / 24, X = rem % 24;
    if ((t >> 7) == 0) {           // channels 0..63 : relu(instnorm(v))
      for (int c = 0; c < 64; ++c) {
        float v = bf2f(vbuf[((size_t)p * 64 + c) * NVOX + vg]);
        v = v * sc[c] + shf[c];
        xin[c][vx] = f2bf(v > 0.f ? v : 0.f);
      }
    } else {                       // channels 64..127 : o1 crop
      int a = (z0 + Z) * 16384 + (y0 + Y) * 128 + (x0 + X);
      for (int c = 0; c < 64; ++c)
        xin[64 + c][vx] = f2bf(out1[(size_t)c * 1048576 + a]);
    }
  }
  __syncthreads();

  const int wave = t >> 5, lane = t & 31;
  const int m = lane & 15, hh = lane >> 4;
  const int mt = wave & 3, ngb = (wave >> 2) * 4;
  float lsum[8] = {}, lsq[8] = {};
  for (int jg = 0; jg < 4; ++jg) {
    int ng = ngb + jg;
    v8f acc = {};
#pragma unroll
    for (int ks = 0; ks < 4; ++ks) {
      int k0 = ks * 32;
      BFrag a, b;
#pragma unroll
      for (int v = 0; v < 8; ++v) a.u[v] = *(const unsigned*)&wbf[mt * 16 + m][k0 + abk(v, hh)];
#pragma unroll
      for (int v = 0; v < 8; ++v) b.u[v] = (unsigned)xin[k0 + abk(v, hh)][ng * 16 + m] | ((unsigned)xin[k0 + abk(v, hh) + 1][ng * 16 + m] << 16);
      acc = wmma_bf16(a, b, acc);
    }
    int vg = vbase + ng * 16 + m;
#pragma unroll
    for (int r = 0; r < 8; ++r) {
      int o = mt * 16 + r + 8 * hh;
      float val = acc[r] + b_back2[o];
      vbuf[((size_t)p * 64 + o) * NVOX + vg] = f2bf(val);   // overwrite in place
      lsum[r] += val; lsq[r] += val * val;
    }
  }
#pragma unroll
  for (int r = 0; r < 8; ++r) {
    int o = mt * 16 + r + 8 * hh;
    atomicAdd(&sstat[o][0], lsum[r]);
    atomicAdd(&sstat[o][1], lsq[r]);
  }
  __syncthreads();
  if (t < 64) {
    atomicAdd(stats2 + (p * 64 + t) * 2 + 0, sstat[t][0]);
    atomicAdd(stats2 + (p * 64 + t) * 2 + 1, sstat[t][1]);
  }
}

// ------------------------------------------- norm + relu + maxpool 24 -> 8
__global__ __launch_bounds__(256) void k_pool(
    const float* __restrict__ g_back2, const float* __restrict__ be_back2,
    const float* __restrict__ stats2,
    const unsigned short* __restrict__ fbuf, unsigned short* __restrict__ crops) {
  __shared__ float sc[64], shf[64];
  const int t = threadIdx.x, p = blockIdx.x;
  if (t < 64) {
    float s = stats2[(p * 64 + t) * 2], sq = stats2[(p * 64 + t) * 2 + 1];
    float mu = s * (1.f / NVOX);
    float var = sq * (1.f / NVOX) - mu * mu;
    float rs = rsqrtf(var + 1e-5f);
    float scale = g_back2[t] * rs;
    sc[t] = scale; shf[t] = be_back2[t] - scale * mu;
  }
  __syncthreads();
  for (int i = t; i < 64 * 512; i += 256) {
    int c = i >> 9, cell = i & 511;
    int dz = cell >> 6, dy = (cell >> 3) & 7, dx = cell & 7;
    const unsigned short* base = fbuf + ((size_t)p * 64 + c) * NVOX;
    float mx = -3.4e38f;
    for (int a = 0; a < 3; ++a)
      for (int b = 0; b < 3; ++b)
        for (int cc = 0; cc < 3; ++cc) {
          float v = bf2f(base[(3 * dz + a) * 576 + (3 * dy + b) * 24 + (3 * dx + cc)]);
          v = v * sc[c] + shf[c];
          v = v > 0.f ? v : 0.f;
          mx = v > mx ? v : mx;
        }
    crops[(size_t)p * 32768 + i] = f2bf(mx);
  }
}

// ---------------------------------------------------- fc1 split-K GEMM
__global__ __launch_bounds__(256) void k_fc1(
    const unsigned short* __restrict__ crops, const float* __restrict__ w_fc1,
    float* __restrict__ x1acc) {
  __shared__ unsigned short wt[1024][16];
#if HAVE_ASYNC
  __shared__ __align__(16) float stg[8][2][256];   // per-wave double buffer (16 KB)
#endif
  const int t = threadIdx.x;
  const int wave = t >> 5, lane = t & 31;
  const int m = lane & 15, hh = lane >> 4;
  const int n0 = (blockIdx.x & 31) * 16;
  const int kb = (blockIdx.x >> 5) * 1024;

#if HAVE_ASYNC
  // Wave-private async staging: wave w owns rows {2w, 2w+1}, 8 sub-chunks of
  // 256 floats, HBM->LDS DMA overlapped with bf16 conversion of prior chunk.
  {
    auto issue = [&](int s) {
      int row = 2 * wave + (s >> 2);
      int kseg = (s & 3) * 256;
      const float* g = w_fc1 + (size_t)(n0 + row) * 32768 + kb + kseg + lane * 8;
      float* l = &stg[wave][s & 1][lane * 8];
      async_copy_b128(g, l);
      async_copy_b128(g + 4, l + 4);
    };
    auto convert = [&](int s) {
      int row = 2 * wave + (s >> 2);
      int kseg = (s & 3) * 256;
      const float* b = &stg[wave][s & 1][lane * 8];
#pragma unroll
      for (int q = 0; q < 8; ++q)
        wt[kseg + lane * 8 + q][row] = f2bf(b[q]);
    };
    issue(0);
    for (int s = 1; s < 8; ++s) {
      fence_lds_reads();       // buffer (s&1) fully consumed before DMA reuse
      issue(s);
      wait_async_le2();        // sub-chunk s-1 landed in LDS
      convert(s - 1);
    }
    wait_async_le0();
    convert(7);
  }
#else
  for (int i = t; i < 16 * 1024; i += 256) {
    int n = i >> 10, kk = i & 1023;
    wt[kk][n] = f2bf(w_fc1[(size_t)(n0 + n) * 32768 + kb + kk]);
  }
#endif
  __syncthreads();

  const int mt = wave & 3, kh = wave >> 2;
  const int p0 = mt * 16;
  v8f acc = {};
  for (int s = 0; s < 16; ++s) {
    int kl = kh * 512 + s * 32;
    BFrag a, b;
#pragma unroll
    for (int v = 0; v < 8; ++v)
      a.u[v] = *(const unsigned*)(crops + (size_t)(p0 + m) * 32768 + kb + kl + abk(v, hh));
#pragma unroll
    for (int v = 0; v < 8; ++v) b.u[v] = (unsigned)wt[kl + abk(v, hh)][m] | ((unsigned)wt[kl + abk(v, hh) + 1][m] << 16);
    acc = wmma_bf16(a, b, acc);
  }
#pragma unroll
  for (int r = 0; r < 8; ++r)
    atomicAdd(x1acc + (p0 + r + 8 * hh) * 512 + n0 + m, acc[r]);
}

__global__ void k_x1(const float* __restrict__ x1acc, const float* __restrict__ b_fc1,
                     unsigned short* __restrict__ x1bf) {
  int i = blockIdx.x * 256 + threadIdx.x;   // 32768 total
  float v = x1acc[i] + b_fc1[i & 511];
  x1bf[i] = f2bf(v > 0.f ? v : 0.f);
}

// ---------------------------------------------------- fc2 + logit/delta heads
__global__ __launch_bounds__(256) void k_head(
    const unsigned short* __restrict__ x1bf,
    const float* __restrict__ w_fc2,  const float* __restrict__ b_fc2,
    const float* __restrict__ w_logit,const float* __restrict__ b_logit,
    const float* __restrict__ w_delta,const float* __restrict__ b_delta,
    float* __restrict__ out) {
  __shared__ unsigned short bt[512][16];
  __shared__ unsigned short x2[64][256];
  const int t = threadIdx.x;
  const int wave = t >> 5, lane = t & 31;
  const int m = lane & 15, hh = lane >> 4;

  for (int nt = 0; nt < 16; ++nt) {     // fc2: N = 256 in 16 tiles
    __syncthreads();
    for (int i = t; i < 16 * 512; i += 256) {
      int n = i >> 9, k = i & 511;
      bt[k][n] = f2bf(w_fc2[(nt * 16 + n) * 512 + k]);
    }
    __syncthreads();
    if (wave < 4) {
      int p0 = wave * 16;
      v8f acc = {};
      for (int s = 0; s < 16; ++s) {
        int k0 = s * 32;
        BFrag a, b;
#pragma unroll
        for (int v = 0; v < 8; ++v)
          a.u[v] = *(const unsigned*)(x1bf + (p0 + m) * 512 + k0 + abk(v, hh));
#pragma unroll
        for (int v = 0; v < 8; ++v) b.u[v] = (unsigned)bt[k0 + abk(v, hh)][m] | ((unsigned)bt[k0 + abk(v, hh) + 1][m] << 16);
        acc = wmma_bf16(a, b, acc);
      }
      int n = nt * 16 + m;
#pragma unroll
      for (int r = 0; r < 8; ++r) {
        float v = acc[r] + b_fc2[n];
        x2[p0 + r + 8 * hh][n] = f2bf(v > 0.f ? v : 0.f);
      }
    }
  }
  __syncthreads();
  for (int i = t; i < 16 * 256; i += 256) {   // padded head weights [256k][16n]
    int n = i >> 8, k = i & 255;
    float v = 0.f;
    if (n < 2)       v = w_logit[n * 256 + k];
    else if (n < 14) v = w_delta[(n - 2) * 256 + k];
    bt[k][n] = f2bf(v);
  }
  __syncthreads();
  if (wave < 4) {
    int p0 = wave * 16;
    v8f acc = {};
    for (int s = 0; s < 8; ++s) {
      int k0 = s * 32;
      BFrag a, b;
#pragma unroll
      for (int v = 0; v < 8; ++v) a.u[v] = *(const unsigned*)&x2[p0 + m][k0 + abk(v, hh)];
#pragma unroll
      for (int v = 0; v < 8; ++v) b.u[v] = (unsigned)bt[k0 + abk(v, hh)][m] | ((unsigned)bt[k0 + abk(v, hh) + 1][m] << 16);
      acc = wmma_bf16(a, b, acc);
    }
    int n = m;
#pragma unroll
    for (int r = 0; r < 8; ++r) {
      int p = p0 + r + 8 * hh;
      if (n < 2)       out[OFF_CL + p * 2 + n] = acc[r] + b_logit[n];
      else if (n < 14) out[OFF_CD + p * 12 + (n - 2)] = acc[r] + b_delta[n - 2];
    }
  }
}

extern "C" void kernel_launch(void* const* d_in, const int* in_sizes, int n_in,
                              void* d_out, int out_size, void* d_ws, size_t ws_size,
                              hipStream_t stream) {
  (void)in_sizes; (void)n_in; (void)out_size; (void)ws_size;
  const float* out1    = (const float*)d_in[0];
  const float* comb2   = (const float*)d_in[1];
  const int*   props   = (const int*)d_in[2];
  const float* w_rpn   = (const float*)d_in[3];
  const float* b_rpn   = (const float*)d_in[4];
  const float* w_rlog  = (const float*)d_in[5];
  const float* b_rlog  = (const float*)d_in[6];
  const float* w_rdel  = (const float*)d_in[7];
  const float* b_rdel  = (const float*)d_in[8];
  const float* w_up2   = (const float*)d_in[9];
  const float* b_up2   = (const float*)d_in[10];
  const float* g_up2   = (const float*)d_in[11];
  const float* be_up2  = (const float*)d_in[12];
  const float* w_back2 = (const float*)d_in[13];
  const float* b_back2 = (const float*)d_in[14];
  const float* g_back2 = (const float*)d_in[15];
  const float* be_back2= (const float*)d_in[16];
  const float* w_fc1   = (const float*)d_in[17];
  const float* b_fc1   = (const float*)d_in[18];
  const float* w_fc2   = (const float*)d_in[19];
  const float* b_fc2   = (const float*)d_in[20];
  const float* w_logit = (const float*)d_in[21];
  const float* b_logit = (const float*)d_in[22];
  const float* w_delta = (const float*)d_in[23];
  const float* b_delta = (const float*)d_in[24];
  float* out = (float*)d_out;

  char* ws = (char*)d_ws;
  float* stats1 = (float*)ws;                          //  8192 f32
  float* stats2 = (float*)(ws + 32768);                //  8192 f32
  float* x1acc  = (float*)(ws + 65536);                // 32768 f32
  unsigned short* x1bf  = (unsigned short*)(ws + 196608);   // 64x512 bf16
  unsigned short* crops = (unsigned short*)(ws + 262144);   // 64x32768 bf16
  unsigned short* vbuf  = (unsigned short*)(ws + 4456448);  // 64x64x13824 bf16

  k_zero<<<192, 256, 0, stream>>>((float*)ws, 49152);
  k_rpn<<<RPN_VOX / 64, 256, 0, stream>>>(comb2, w_rpn, b_rpn, w_rlog, b_rlog,
                                          w_rdel, b_rdel, out);
  k_up2<<<64 * NCHUNK, 256, 0, stream>>>(comb2, props, w_up2, b_up2, vbuf, stats1);
  k_back2<<<64 * NCHUNK, 256, 0, stream>>>(out1, props, w_back2, b_back2, g_up2,
                                           be_up2, stats1, vbuf, stats2);
  k_pool<<<64, 256, 0, stream>>>(g_back2, be_back2, stats2, vbuf, crops);
  k_fc1<<<1024, 256, 0, stream>>>(crops, w_fc1, x1acc);
  k_x1<<<128, 256, 0, stream>>>(x1acc, b_fc1, x1bf);
  k_head<<<1, 256, 0, stream>>>(x1bf, w_fc2, b_fc2, w_logit, b_logit,
                                w_delta, b_delta, out);
}
